// CPAMDec_Mix_55439437857293
// MI455X (gfx1250) — matrix operand order repaired and checked by reading, not compiled, not run on hardware
//
#include <hip/hip_runtime.h>

// ---------------------------------------------------------------------------
// CPAM dual-attention mix, fused for MI455X (gfx1250, wave32, WMMA f16->f32)
// ---------------------------------------------------------------------------

typedef __attribute__((ext_vector_type(16))) _Float16 v16h;
typedef __attribute__((ext_vector_type(8)))  float    v8f;

union Frag16 { v16h h; float4 q[2]; };

__device__ __forceinline__ v16h load_frag(const void* p0, const void* p1) {
  Frag16 f;
  f.q[0] = *(const float4*)p0;
  f.q[1] = *(const float4*)p1;
  return f.h;
}

// ---------------------------------------------------------------------------
// Kernel 1: convert wq1/wq2 (D x C, f32) -> f16 (row-major, K=C contiguous)
// ---------------------------------------------------------------------------
__global__ __launch_bounds__(256)
void cpam_cvt(const float* __restrict__ wq1, const float* __restrict__ wq2,
              _Float16* __restrict__ wq1h, _Float16* __restrict__ wq2h) {
  int i = blockIdx.x * 256 + threadIdx.x;   // 128*512 = 65536 elements each
  wq1h[i] = (_Float16)wq1[i];
  wq2h[i] = (_Float16)wq2[i];
}

// ---------------------------------------------------------------------------
// Kernel 2: k = y @ wk^T + bk  (N,K,D) f16 row-major (D contiguous)
//           v = y @ wv^T + bv  stored TRANSPOSED as (N,C,K) f16 (K contiguous)
// One block per (n, kk); tiny GEMMs, plain f32 FMA is plenty.
// ---------------------------------------------------------------------------
__global__ __launch_bounds__(256)
void cpam_kv(const float* __restrict__ y1, const float* __restrict__ y2,
             const float* __restrict__ wk1, const float* __restrict__ bk1,
             const float* __restrict__ wk2, const float* __restrict__ bk2,
             const float* __restrict__ wv1, const float* __restrict__ bv1,
             const float* __restrict__ wv2, const float* __restrict__ bv2,
             _Float16* __restrict__ k1h, _Float16* __restrict__ k2h,
             _Float16* __restrict__ v1T, _Float16* __restrict__ v2T) {
  const int n  = blockIdx.x >> 6;
  const int kk = blockIdx.x & 63;
  const int tid = threadIdx.x;
  __shared__ float sy[2][512];
  {
    const float* y1r = y1 + ((size_t)n * 64 + kk) * 512;
    const float* y2r = y2 + ((size_t)n * 64 + kk) * 512;
    sy[0][tid] = y1r[tid]; sy[0][tid + 256] = y1r[tid + 256];
    sy[1][tid] = y2r[tid]; sy[1][tid + 256] = y2r[tid + 256];
  }
  __syncthreads();
  // keys: threads 0..127 -> k1[d], 128..255 -> k2[d]
  {
    const int d = tid & 127;
    const int which = tid >> 7;
    const float* w  = (which ? wk2 : wk1) + (size_t)d * 512;
    const float* yy = sy[which];
    float acc = which ? bk2[d] : bk1[d];
    for (int c = 0; c < 512; c += 4) {
      float4 wv = *(const float4*)(w + c);
      acc += yy[c] * wv.x + yy[c + 1] * wv.y + yy[c + 2] * wv.z + yy[c + 3] * wv.w;
    }
    (which ? k2h : k1h)[((size_t)n * 64 + kk) * 128 + d] = (_Float16)acc;
  }
  // values (stored transposed: [n][c][kk])
  for (int c = tid; c < 512; c += 256) {
    float a1 = bv1[c], a2 = bv2[c];
    const float* w1 = wv1 + (size_t)c * 512;
    const float* w2 = wv2 + (size_t)c * 512;
    for (int j = 0; j < 512; j += 4) {
      float4 q1 = *(const float4*)(w1 + j);
      float4 q2 = *(const float4*)(w2 + j);
      a1 += sy[0][j] * q1.x + sy[0][j + 1] * q1.y + sy[0][j + 2] * q1.z + sy[0][j + 3] * q1.w;
      a2 += sy[1][j] * q2.x + sy[1][j + 1] * q2.y + sy[1][j + 2] * q2.z + sy[1][j + 3] * q2.w;
    }
    v1T[((size_t)n * 512 + c) * 64 + kk] = (_Float16)a1;
    v2T[((size_t)n * 512 + c) * 64 + kk] = (_Float16)a2;
  }
}

// ---------------------------------------------------------------------------
// Kernel 3: fused main pass. One block = one (n, 64-pixel tile).
//   q = x^T Wq^T + b  (WMMA, f16 in / f32 acc)
//   e = |q1 k1^T - q2 k2^T| ; softmax over K=64
//   out = scale * (attn @ v) + x   (transposed through LDS, coalesced stores)
// ---------------------------------------------------------------------------
__global__ __launch_bounds__(256)
void cpam_main(const float* __restrict__ x1, const float* __restrict__ x2,
               const float* __restrict__ scale,
               const float* __restrict__ bq1, const float* __restrict__ bq2,
               const _Float16* __restrict__ wq1h, const _Float16* __restrict__ wq2h,
               const _Float16* __restrict__ k1h, const _Float16* __restrict__ k2h,
               const _Float16* __restrict__ v1T, const _Float16* __restrict__ v2T,
               float* __restrict__ out) {
  constexpr int C = 512, HW = 4096, K = 64, D = 128;
  // Overlaid LDS arena (phases never alias live data; barriers between phases)
  __shared__ __align__(16) char smem[52224];
  _Float16* sQ1    = (_Float16*)(smem);            // 64 rows x stride 136 halfs
  _Float16* sQ2    = (_Float16*)(smem + 17408);
  _Float16* sStage = (_Float16*)(smem + 34816);    // 64 rows x stride 72 halfs
  float*    sE     = (float*)(smem + 34816);       // 64 rows x stride 68 f32
  _Float16* sAttn  = (_Float16*)(smem);            // 64 rows x stride 72 halfs
  float*    sX     = (float*)(smem + 17408);       // 64 rows x stride 132 f32

  const int tid  = threadIdx.x;
  const int lane = tid & 31;
  const int wid  = tid >> 5;
  const int mt   = wid & 3;            // 16-row tile of this wave
  const int nh   = wid >> 2;           // which 64-col half of this wave
  const int ln16 = lane & 15;
  const int sel  = lane >> 4;          // 0/1: K-half select in fragments
  const int rowA = mt * 16 + ln16;     // A-matrix row owned by this lane

  const int n     = blockIdx.x >> 6;
  const int lbase = (blockIdx.x & 63) * 64;
  const float scl = scale[0];

  // ---- Phase 1: q1, q2 projections (64 x 128 each) -------------------------
  for (int term = 0; term < 2; ++term) {
    const float* xp = (term ? x2 : x1) + (size_t)n * C * HW;
    const _Float16* wq = term ? wq2h : wq1h;
    const float* bq = term ? bq2 : bq1;
    _Float16* sQ = term ? sQ2 : sQ1;

    v8f acc[4] = {};
    for (int s = 0; s < 8; ++s) {               // 8 stages x 64 C-cols
      const int cbase = s * 64;
      #pragma unroll
      for (int i = 0; i < 16; ++i) {            // stage 64l x 64c f32->f16
        int e = tid + 256 * i;
        int cl = e >> 6, l = e & 63;
        sStage[l * 72 + cl] = (_Float16)xp[(size_t)(cbase + cl) * HW + lbase + l];
      }
      __syncthreads();
      #pragma unroll
      for (int ks = 0; ks < 2; ++ks) {          // two K=32 steps per stage
        const char* ap = (const char*)sStage + rowA * 144 + ks * 64 + sel * 16;
        v16h afrag = load_frag(ap, ap + 32);
        const int kglob = cbase + ks * 32;
        #pragma unroll
        for (int nt = 0; nt < 4; ++nt) {
          const int col = nh * 64 + nt * 16 + ln16;
          const _Float16* bp = wq + (size_t)col * C + kglob + sel * 8;
          v16h bfrag = load_frag(bp, bp + 16);
          acc[nt] = __builtin_amdgcn_wmma_f32_16x16x32_f16(
              false, afrag, false, bfrag, (short)0, acc[nt], false, false);
        }
      }
      __syncthreads();
    }
    #pragma unroll
    for (int nt = 0; nt < 4; ++nt) {            // bias + f16 store to sQ
      const int col = nh * 64 + nt * 16 + ln16;
      const float b = bq[col];
      #pragma unroll
      for (int r = 0; r < 8; ++r) {
        const int row = mt * 16 + r + sel * 8;
        sQ[row * 136 + col] = (_Float16)(acc[nt][r] + b);
      }
    }
  }
  __syncthreads();

  // ---- Phase 2: energy = |q1 k1^T - q2 k2^T|  (64 x 64) --------------------
  {
    v8f e1[2] = {}, e2[2] = {};
    const int nt0 = nh * 2;                     // 2 of 4 col-tiles per wave
    #pragma unroll
    for (int kt = 0; kt < 4; ++kt) {            // D=128 -> 4 K-steps
      const char* a1p = (const char*)sQ1 + rowA * 272 + kt * 64 + sel * 16;
      const char* a2p = (const char*)sQ2 + rowA * 272 + kt * 64 + sel * 16;
      v16h a1 = load_frag(a1p, a1p + 32);
      v16h a2 = load_frag(a2p, a2p + 32);
      #pragma unroll
      for (int t = 0; t < 2; ++t) {
        const int kcol = (nt0 + t) * 16 + ln16;
        const _Float16* b1p = k1h + ((size_t)n * K + kcol) * D + kt * 32 + sel * 8;
        const _Float16* b2p = k2h + ((size_t)n * K + kcol) * D + kt * 32 + sel * 8;
        v16h b1 = load_frag(b1p, b1p + 16);
        v16h b2 = load_frag(b2p, b2p + 16);
        e1[t] = __builtin_amdgcn_wmma_f32_16x16x32_f16(false, a1, false, b1, (short)0, e1[t], false, false);
        e2[t] = __builtin_amdgcn_wmma_f32_16x16x32_f16(false, a2, false, b2, (short)0, e2[t], false, false);
      }
    }
    #pragma unroll
    for (int t = 0; t < 2; ++t) {
      const int col = (nt0 + t) * 16 + ln16;
      #pragma unroll
      for (int r = 0; r < 8; ++r)
        sE[(mt * 16 + r + sel * 8) * 68 + col] = fabsf(e1[t][r] - e2[t][r]);
    }
  }
  __syncthreads();

  // ---- Phase 3: softmax over K=64 per pixel row ----------------------------
  if (tid < 64) {
    const float* er = sE + tid * 68;
    float m = er[0];
    for (int j = 1; j < 64; ++j) m = fmaxf(m, er[j]);
    float sum = 0.f;
    for (int j = 0; j < 64; ++j) sum += __expf(er[j] - m);
    const float inv = 1.f / sum;
    for (int j = 0; j < 64; ++j)
      sAttn[tid * 72 + j] = (_Float16)(__expf(er[j] - m) * inv);
  }
  __syncthreads();

  // ---- Phase 4: out = scale * (attn @ v) + x, transpose via LDS ------------
  for (int term = 0; term < 2; ++term) {
    const _Float16* vT = term ? v2T : v1T;
    const float* xp = (term ? x2 : x1) + (size_t)n * C * HW;
    float* op = out + (size_t)term * (8ll * C * HW) + (size_t)n * C * HW;
    for (int cc = 0; cc < 4; ++cc) {            // 4 chunks of 128 C-cols
      const int cbase = cc * 128;
      v8f acc[4] = {};
      #pragma unroll
      for (int kt = 0; kt < 2; ++kt) {          // K=64 -> 2 K-steps
        const char* ap = (const char*)sAttn + rowA * 144 + kt * 64 + sel * 16;
        v16h a = load_frag(ap, ap + 32);
        #pragma unroll
        for (int nt = 0; nt < 4; ++nt) {
          const int ccol = nh * 64 + nt * 16 + ln16;       // 0..127 in chunk
          const _Float16* bp = vT + ((size_t)n * C + cbase + ccol) * K + kt * 32 + sel * 8;
          v16h b = load_frag(bp, bp + 16);
          acc[nt] = __builtin_amdgcn_wmma_f32_16x16x32_f16(
              false, a, false, b, (short)0, acc[nt], false, false);
        }
      }
      #pragma unroll
      for (int nt = 0; nt < 4; ++nt) {          // stash f32 for transpose
        const int ccol = nh * 64 + nt * 16 + ln16;
        #pragma unroll
        for (int r = 0; r < 8; ++r)
          sX[(mt * 16 + r + sel * 8) * 132 + ccol] = acc[nt][r];
      }
      __syncthreads();
      #pragma unroll
      for (int i = 0; i < 32; ++i) {            // coalesced store + residual
        int e = tid + 256 * i;
        int cl = e >> 6, l = e & 63;
        size_t gidx = (size_t)(cbase + cl) * HW + lbase + l;
        op[gidx] = scl * sX[l * 132 + cl] + xp[gidx];
      }
      __syncthreads();
    }
  }
}

// ---------------------------------------------------------------------------
extern "C" void kernel_launch(void* const* d_in, const int* in_sizes, int n_in,
                              void* d_out, int out_size, void* d_ws, size_t ws_size,
                              hipStream_t stream) {
  (void)in_sizes; (void)n_in; (void)out_size; (void)ws_size;
  const float* x1  = (const float*)d_in[0];
  const float* y1  = (const float*)d_in[1];
  const float* x2  = (const float*)d_in[2];
  const float* y2  = (const float*)d_in[3];
  const float* scl = (const float*)d_in[4];
  const float* wq1 = (const float*)d_in[5];
  const float* bq1 = (const float*)d_in[6];
  const float* wk1 = (const float*)d_in[7];
  const float* bk1 = (const float*)d_in[8];
  const float* wv1 = (const float*)d_in[9];
  const float* bv1 = (const float*)d_in[10];
  const float* wq2 = (const float*)d_in[11];
  const float* bq2 = (const float*)d_in[12];
  const float* wk2 = (const float*)d_in[13];
  const float* bk2 = (const float*)d_in[14];
  const float* wv2 = (const float*)d_in[15];
  const float* bv2 = (const float*)d_in[16];

  // workspace layout (bytes): wq1h wq2h k1h k2h v1T v2T  (total ~1.5 MB)
  char* ws = (char*)d_ws;
  _Float16* wq1h = (_Float16*)(ws);
  _Float16* wq2h = (_Float16*)(ws + 131072);
  _Float16* k1h  = (_Float16*)(ws + 262144);
  _Float16* k2h  = (_Float16*)(ws + 393216);
  _Float16* v1T  = (_Float16*)(ws + 524288);
  _Float16* v2T  = (_Float16*)(ws + 1048576);

  cpam_cvt<<<256, 256, 0, stream>>>(wq1, wq2, wq1h, wq2h);
  cpam_kv<<<512, 256, 0, stream>>>(y1, y2, wk1, bk1, wk2, bk2,
                                   wv1, bv1, wv2, bv2, k1h, k2h, v1T, v2T);
  cpam_main<<<512, 256, 0, stream>>>(x1, x2, scl, bq1, bq2, wq1h, wq2h,
                                     k1h, k2h, v1T, v2T, (float*)d_out);
}
